// re_unit_14869176778870
// MI455X (gfx1250) — matrix operand order repaired and checked by reading, not compiled, not run on hardware
//
#include <hip/hip_runtime.h>

typedef _Float16 v16h  __attribute__((ext_vector_type(16)));
typedef _Float16 v8h   __attribute__((ext_vector_type(8)));
typedef _Float16 v4h   __attribute__((ext_vector_type(4)));
typedef float    v8f   __attribute__((ext_vector_type(8)));
typedef float    f32x4 __attribute__((ext_vector_type(4)));
typedef unsigned int u32x4 __attribute__((ext_vector_type(4)));

#define Lq 128
#define Bq 8
#define Hq 256
#define Rq 97
#define KP 264   // padded LDS row stride in halves (528 B -> bank-quad rotation)

// ---- WMMA fragment loaders (f16 16x16x32; contiguous k = 16h+s in both A and B, not ISA table order) ----
// A: lanes 0-15 row=lane hold K 0..15; lanes 16-31 hold K 16..31
static __device__ inline v16h frag_a(const _Float16* p) {
  v8h lo = *(const v8h*)p;          // 8 halves, contiguous
  v8h hi = *(const v8h*)(p + 8);    // +16B
  return __builtin_shufflevector(lo, hi, 0,1,2,3,4,5,6,7,8,9,10,11,12,13,14,15);
}
// B (column-major staged): lanes 0-15 col=lane hold K 0..15; lanes 16-31 K 16..31
static __device__ inline v16h frag_b(const _Float16* p) {
  v8h lo = *(const v8h*)p;
  v8h hi = *(const v8h*)(p + 8);
  return __builtin_shufflevector(lo, hi, 0,1,2,3,4,5,6,7,8,9,10,11,12,13,14,15);
}

static __device__ inline v8f wmma_f16(v16h a, v16h b, v8f c) {
  return __builtin_amdgcn_wmma_f32_16x16x32_f16(false, a, false, b, (short)0, c,
                                                false, false);
}

// ---------------- prep: f16 transposed weights ----------------
__global__ __launch_bounds__(256) void k_prep_w(const float* __restrict__ W_hid,
                                                const float* __restrict__ W_rel,
                                                _Float16* __restrict__ W3T,
                                                _Float16* __restrict__ WrT) {
  int idx = blockIdx.x * 256 + threadIdx.x;   // 0..65535 = n*256 + k
  int n = idx >> 8, k = idx & 255;
  W3T[idx] = (_Float16)W_hid[(2 * Hq + k) * Hq + n];       // W3T[n][k] = W3[k][n]
  if (n < 112)
    WrT[idx] = (n < Rq) ? (_Float16)W_rel[k * Rq + n] : (_Float16)0.f;
}

// ---------------- prep: t1 = h_re@W1, t2 = h_re@W2 (f32) ----------------
__global__ __launch_bounds__(256) void k_t12(const float* __restrict__ h_re,
                                             const float* __restrict__ W_hid,
                                             float* __restrict__ t1,
                                             float* __restrict__ t2) {
  __shared__ float sh[8 * 256];
  const int i = blockIdx.x;       // 0..127
  const int n = threadIdx.x;      // 0..255
  #pragma unroll
  for (int b = 0; b < 8; ++b) sh[b * 256 + n] = h_re[(i * Bq + b) * Hq + n];
  __syncthreads();
  float a1[8] = {}, a2[8] = {};
  for (int k = 0; k < 256; ++k) {
    float w1 = W_hid[k * Hq + n];
    float w2 = W_hid[(Hq + k) * Hq + n];
    #pragma unroll
    for (int b = 0; b < 8; ++b) {
      float h = sh[b * 256 + k];
      a1[b] += h * w1;
      a2[b] += h * w2;
    }
  }
  #pragma unroll
  for (int b = 0; b < 8; ++b) {
    t1[(i * Bq + b) * Hq + n] = a1[b];
    t2[(i * Bq + b) * Hq + n] = a2[b];
  }
}

// ---------------- fused main kernel ----------------
__global__ __launch_bounds__(256) void k_fused(
    const float* __restrict__ h_share, const float* __restrict__ mask,
    const float* __restrict__ b_hid, const float* __restrict__ ln_g,
    const float* __restrict__ ln_b, const float* __restrict__ b_rel,
    const float* __restrict__ t1, const float* __restrict__ t2,
    const _Float16* __restrict__ W3T, const _Float16* __restrict__ WrT,
    float* __restrict__ out) {
  __shared__ __attribute__((aligned(16))) _Float16 sA[128 * KP]; // A tile, then X
  __shared__ __attribute__((aligned(16))) _Float16 sB[256 * KP]; // W3T, then WrT
  __shared__ float sS[128 * 2];   // per-row partial sums  (by N-half)
  __shared__ float sQ[128 * 2];   // per-row partial sumsq (by N-half)

  const int tid = threadIdx.x;
  const int blk = blockIdx.x;   // 0..1023; rows blk*128 .. +127 of (i,j,b)

  // ---- stage A tile: 128 rows x 256 K, f32 -> f16, coalesced ----
  {
    const f32x4* src = (const f32x4*)(h_share + (size_t)blk * (128 * 256));
    #pragma unroll
    for (int it = 0; it < 32; ++it) {
      int e   = it * 256 + tid;      // float4 index 0..8191
      f32x4 f = src[e];
      int row = e >> 6;              // 64 float4 per row
      int c   = (e & 63) << 2;       // half offset
      v4h p = { (_Float16)f.x, (_Float16)f.y, (_Float16)f.z, (_Float16)f.w };
      *(v4h*)&sA[row * KP + c] = p;
    }
  }
  // ---- stage B = W3T (256 x 256 f16) ----
  {
    const u32x4* src = (const u32x4*)W3T;
    #pragma unroll
    for (int it = 0; it < 32; ++it) {
      int e   = it * 256 + tid;      // uint4 index 0..8191
      int row = e >> 5;              // 32 uint4 per row
      int c   = (e & 31) << 3;       // half offset
      *(u32x4*)&sB[row * KP + c] = src[e];
    }
  }
  __syncthreads();

  const int wave = tid >> 5;
  const int lane = tid & 31;
  const int lx   = lane & 15;
  const int lh   = lane >> 4;
  const int mbase = (wave >> 1) * 32;   // 32 rows per wave pair
  const int ng    = wave & 1;           // N half: cols ng*128 .. +127
  const int nbase = ng * 128;

  v8f zero = {};
  v8f acc[2][8];
  #pragma unroll
  for (int u = 0; u < 2; ++u)
    #pragma unroll
    for (int t = 0; t < 8; ++t) acc[u][t] = zero;

  // ---- GEMM1: 32 rows x 128 cols per wave, K=256; B double-buffered ----
  #pragma unroll
  for (int kk = 0; kk < 256; kk += 32) {
    const _Float16* arow = &sA[(mbase + lx) * KP + kk + (lh << 4)];
    v16h a0 = frag_a(arow);
    v16h a1 = frag_a(arow + 16 * KP);
    v16h bc = frag_b(&sB[(nbase + lx) * KP + kk + (lh << 4)]);
    #pragma unroll
    for (int t = 0; t < 8; ++t) {
      v16h bn = bc;
      if (t < 7)
        bn = frag_b(&sB[(nbase + (t + 1) * 16 + lx) * KP + kk + (lh << 4)]);
      acc[0][t] = wmma_f16(a0, bc, acc[0][t]);
      acc[1][t] = wmma_f16(a1, bc, acc[1][t]);
      bc = bn;
    }
  }

  // ---- epilogue 1a: += t1[i,b] + t2[j,b] + b_hid; partial row stats ----
  const int i_idx = blk >> 3;        // 1024 rows per i, 128 rows per block
  float bh[8];
  #pragma unroll
  for (int t = 0; t < 8; ++t) bh[t] = b_hid[nbase + t * 16 + lx];

  #pragma unroll
  for (int u = 0; u < 2; ++u) {
    #pragma unroll
    for (int v = 0; v < 8; ++v) {
      int rl  = mbase + u * 16 + (lh << 3) + v;   // local row 0..127
      int gr  = blk * 128 + rl;
      int rem = gr & 1023;
      int j   = rem >> 3;
      int b   = rem & 7;
      const float* p1 = t1 + (i_idx * Bq + b) * Hq + nbase;
      const float* p2 = t2 + (j * Bq + b) * Hq + nbase;
      float s = 0.f, q = 0.f;
      #pragma unroll
      for (int t = 0; t < 8; ++t) {
        int cl  = t * 16 + lx;
        float x = acc[u][t][v] + p1[cl] + p2[cl] + bh[t];
        acc[u][t][v] = x;
        s += x;
        q += x * x;
      }
      #pragma unroll
      for (int m = 1; m <= 8; m <<= 1) {
        s += __shfl_xor(s, m, 32);
        q += __shfl_xor(q, m, 32);
      }
      if (lx == 0) {            // lanes 0 and 16 publish their rows
        sS[rl * 2 + ng] = s;
        sQ[rl * 2 + ng] = q;
      }
    }
  }
  __syncthreads();   // partial stats visible; also: all waves done with GEMM1

  // ---- epilogue 1b: LayerNorm + ELU, write X (f16) back into sA ----
  float ga[8], be[8];
  #pragma unroll
  for (int t = 0; t < 8; ++t) {
    ga[t] = ln_g[nbase + t * 16 + lx];
    be[t] = ln_b[nbase + t * 16 + lx];
  }
  #pragma unroll
  for (int u = 0; u < 2; ++u) {
    #pragma unroll
    for (int v = 0; v < 8; ++v) {
      int rl = mbase + u * 16 + (lh << 3) + v;
      float mu  = (sS[rl * 2] + sS[rl * 2 + 1]) * (1.f / 256.f);
      float var = (sQ[rl * 2] + sQ[rl * 2 + 1]) * (1.f / 256.f) - mu * mu;
      float rs  = rsqrtf(var + 1e-5f);
      #pragma unroll
      for (int t = 0; t < 8; ++t) {
        float x = (acc[u][t][v] - mu) * rs * ga[t] + be[t];
        x = (x > 0.f) ? x : (__expf(x) - 1.f);   // ELU(alpha=1)
        sA[rl * KP + nbase + t * 16 + lx] = (_Float16)x;
      }
    }
  }

  __syncthreads();   // X complete in sA; all waves done reading W3T from sB
  // ---- stage B2 = WrT (112 x 256 f16) into sB ----
  {
    const u32x4* src = (const u32x4*)WrT;
    #pragma unroll
    for (int it = 0; it < 14; ++it) {
      int e   = it * 256 + tid;      // 0..3583
      int row = e >> 5;
      int c   = (e & 31) << 3;
      *(u32x4*)&sB[row * KP + c] = src[e];
    }
  }
  __syncthreads();

  // ---- GEMM2: 16 rows x 112 cols (97 valid) per wave, K=256 ----
  const int m0 = wave * 16;
  v8f acc2[7];
  #pragma unroll
  for (int t = 0; t < 7; ++t) acc2[t] = zero;
  #pragma unroll
  for (int kk = 0; kk < 256; kk += 32) {
    const v16h a = frag_a(&sA[(m0 + lx) * KP + kk + (lh << 4)]);
    v16h bc = frag_b(&sB[lx * KP + kk + (lh << 4)]);
    #pragma unroll
    for (int t = 0; t < 7; ++t) {
      v16h bn = bc;
      if (t < 6)
        bn = frag_b(&sB[((t + 1) * 16 + lx) * KP + kk + (lh << 4)]);
      acc2[t] = wmma_f16(a, bc, acc2[t]);
      bc = bn;
    }
  }

  // ---- epilogue 2: sigmoid(+b_rel) * mask_i * mask_j, store f32 ----
  #pragma unroll
  for (int v = 0; v < 8; ++v) {
    int gr  = blk * 128 + m0 + (lh << 3) + v;
    int rem = gr & 1023;
    int j   = rem >> 3;
    int b   = rem & 7;
    float mm  = mask[i_idx * Bq + b] * mask[j * Bq + b];
    float* po = out + (size_t)gr * Rq;
    #pragma unroll
    for (int t = 0; t < 7; ++t) {
      int r = t * 16 + lx;
      if (r < Rq) {
        float x  = acc2[t][v] + b_rel[r];
        float sg = 1.f / (1.f + __expf(-x));
        po[r] = sg * mm;
      }
    }
  }
}

extern "C" void kernel_launch(void* const* d_in, const int* in_sizes, int n_in,
                              void* d_out, int out_size, void* d_ws, size_t ws_size,
                              hipStream_t stream) {
  const float* h_re    = (const float*)d_in[0];
  const float* h_share = (const float*)d_in[1];
  const float* mask    = (const float*)d_in[2];
  const float* W_hid   = (const float*)d_in[3];
  const float* b_hid   = (const float*)d_in[4];
  const float* ln_g    = (const float*)d_in[5];
  const float* ln_b    = (const float*)d_in[6];
  const float* W_rel   = (const float*)d_in[7];
  const float* b_rel   = (const float*)d_in[8];

  char* ws = (char*)d_ws;
  float*    t1  = (float*)ws;                              // 1 MB
  float*    t2  = (float*)(ws + 1048576);                  // 1 MB
  _Float16* W3T = (_Float16*)(ws + 2097152);               // 128 KB
  _Float16* WrT = (_Float16*)(ws + 2097152 + 131072);      // 56 KB

  k_prep_w<<<256, 256, 0, stream>>>(W_hid, W_rel, W3T, WrT);
  k_t12<<<128, 256, 0, stream>>>(h_re, W_hid, t1, t2);
  k_fused<<<1024, 256, 0, stream>>>(h_share, mask, b_hid, ln_g, ln_b, b_rel,
                                    t1, t2, W3T, WrT, (float*)d_out);
}
